// YoloHead_59854664237285
// MI455X (gfx1250) — matrix operand order repaired and checked by reading, not compiled, not run on hardware
//
#include <hip/hip_runtime.h>
#include <math.h>

// ---------------- problem constants (match reference) ----------------
#define NC_   80
#define NA_   3
#define NO_   85        // NC + 5
#define NT_   200
#define BS_   8
#define INCH  256
#define ROWS_PER_A 1000 // 5 offsets * NT
#define ROWS_LVL   3000 // NA * 5 * NT
#define META_STRIDE 12
#define PS_STRIDE   96  // NO_ padded
#define NTILES_O   6    // ceil(85/16)

// ---------------- workspace layout (float units) ----------------
#define WS_OBJ    0          // 201600 obj logits  (lvl offsets 0,153600,192000)
#define WS_TOBJ   201600     // 201600 uint bit patterns (tobj)
#define WS_META   403200     // 3 * 3000 * 12
#define WS_PS     511200     // 3 * 3000 * 96
#define WS_NVAL   1375200    // 3
#define WS_LBOXP  1375232    // 3 * 16
#define WS_LCLSP  1375280    // 3 * 16
#define WS_FGP    1375328    // 3 * 128
#define WS_BGP    1375712    // 3 * 128
#define WS_BCEP   1376096    // 3 * 128
#define WS_BF_OBJ 1376480    // 3 * 2048  (bf16 fragments, 4096 ushort/level)
#define WS_BF_GAT 1382624    // 3 * 36864 (bf16 fragments, 73728 ushort/level)
#define WS_TOTAL  1493216

typedef __attribute__((ext_vector_type(16))) __bf16        v16bf;
typedef __attribute__((ext_vector_type(8)))  float         v8f;
typedef __attribute__((ext_vector_type(4)))  unsigned int  v4u;

// anchors in grid units are identical across levels: (32/8, 64/16, 128/32) = 4
__constant__ float ANC_W[3] = {5.65685424f, 4.0f, 2.82842712f};
__constant__ float ANC_H[3] = {2.82842712f, 4.0f, 5.65685424f};

__device__ __forceinline__ unsigned short f2bf(float f) {
    unsigned u = __float_as_uint(f);
    unsigned r = u + 0x7FFFu + ((u >> 16) & 1u);   // round-to-nearest-even
    return (unsigned short)(r >> 16);
}
__device__ __forceinline__ float dsigmoid(float x) { return 1.0f / (1.0f + __expf(-x)); }
__device__ __forceinline__ float dsoftplus(float x) {
    return fmaxf(x, 0.0f) + log1pf(__expf(-fabsf(x)));
}

// ---------------- kernel 0: zero tobj + n_valid ----------------
__global__ void k_zero(float* ws) {
    int i = blockIdx.x * blockDim.x + threadIdx.x;
    int stride = gridDim.x * blockDim.x;
    for (int k = i; k < 201600; k += stride) ws[WS_TOBJ + k] = 0.0f;
    if (i < 3) ws[WS_NVAL + i] = 0.0f;
}

// ---------------- B-fragment packers (f32 weights -> per-lane bf16 frags) -------
// fragment layout: frag[((tile*8 + kstep)*32 + lane)*16 + 2*vv(+1)]
// lane<16: n=lane, K pairs {0,2,4,6,16,18,20,22}; lane>=16: n=lane-16, same +8  (B 32x16 bf16)
__global__ void k_pack_obj_b(const float* __restrict__ w, unsigned short* __restrict__ frag) {
    int tid = threadIdx.x;              // 256 = 8 ksteps * 32 lanes
    int lane = tid & 31, ks = tid >> 5;
    int half = lane >> 4, n = lane & 15;
    int k0 = ks * 32;
    unsigned short* out = frag + (size_t)(ks * 32 + lane) * 16;
    #pragma unroll
    for (int vv = 0; vv < 8; ++vv) {
        int kb = (vv < 4 ? 2 * vv : 16 + 2 * (vv - 4)) + (half ? 8 : 0);
        float lo = 0.0f, hi = 0.0f;
        if (n < NA_) {
            const float* bc = w + (size_t)(n * NO_ + 4) * INCH + k0 + kb;
            lo = bc[0]; hi = bc[1];
        }
        out[2 * vv]     = f2bf(lo);
        out[2 * vv + 1] = f2bf(hi);
    }
}

__global__ void k_pack_gat_b(const float* __restrict__ w, unsigned short* __restrict__ frag) {
    int tid = blockIdx.x * blockDim.x + threadIdx.x;   // 3*6*8*32 = 4608
    if (tid >= NA_ * NTILES_O * 8 * 32) return;
    int lane = tid & 31;
    int ks   = (tid >> 5) & 7;
    int nt   = (tid >> 8) % NTILES_O;
    int a    = tid / (NTILES_O * 8 * 32);
    int half = lane >> 4, n = lane & 15;
    int o = nt * 16 + n;
    int k0 = ks * 32;
    unsigned short* out = frag + (size_t)(((a * NTILES_O + nt) * 8 + ks) * 32 + lane) * 16;
    #pragma unroll
    for (int vv = 0; vv < 8; ++vv) {
        int kb = (vv < 4 ? 2 * vv : 16 + 2 * (vv - 4)) + (half ? 8 : 0);
        float lo = 0.0f, hi = 0.0f;
        if (o < NO_) {
            const float* bc = w + (size_t)(a * NO_ + o) * INCH + k0 + kb;
            lo = bc[0]; hi = bc[1];
        }
        out[2 * vv]     = f2bf(lo);
        out[2 * vv + 1] = f2bf(hi);
    }
}

// ---------------- kernel 1: build targets (per level) ----------------
// row order within a level: a*1000 + j*200 + t   (j = offset index 0..4)
__global__ void k_build_targets(const float* __restrict__ tg, float* __restrict__ meta,
                                float* __restrict__ nval, int nx) {
    const float G = 0.5f, AT = 2.91f;
    const float OFFX[5] = {0.f, 0.5f, 0.f, -0.5f, 0.f};
    const float OFFY[5] = {0.f, 0.f, 0.5f, 0.f, -0.5f};
    int pair = blockIdx.x * blockDim.x + threadIdx.x;   // (a,t), 600 total
    float cnt = 0.0f;
    if (pair < NA_ * NT_) {
        int a = pair / NT_, t = pair % NT_;
        float fb  = tg[t * 6 + 0];
        float fc  = tg[t * 6 + 1];
        float gx  = tg[t * 6 + 2] * (float)nx;
        float gy  = tg[t * 6 + 3] * (float)nx;   // ny == nx
        float gw  = tg[t * 6 + 4] * (float)nx;
        float gh  = tg[t * 6 + 5] * (float)nx;
        float rw = gw / ANC_W[a], rh = gh / ANC_H[a];
        float mr = fmaxf(fmaxf(rw, 1.0f / rw), fmaxf(rh, 1.0f / rh));
        bool m_anchor = mr < AT;
        float ix = (float)nx - gx, iy = (float)nx - gy;
        bool nearx = (fmodf(gx, 1.0f) < G) && (gx > 1.0f);
        bool neary = (fmodf(gy, 1.0f) < G) && (gy > 1.0f);
        bool farx  = (fmodf(ix, 1.0f) < G) && (ix > 1.0f);
        bool fary  = (fmodf(iy, 1.0f) < G) && (iy > 1.0f);
        bool sel[5] = {true, nearx, neary, farx, fary};
        #pragma unroll
        for (int j = 0; j < 5; ++j) {
            bool mask = sel[j] && m_anchor;
            int gii = (int)(gx - OFFX[j]);   // trunc like astype(int32)
            int gjj = (int)(gy - OFFY[j]);
            int gic = min(max(gii, 0), nx - 1);
            int gjc = min(max(gjj, 0), nx - 1);
            float* mrow = meta + (size_t)(a * ROWS_PER_A + j * NT_ + t) * META_STRIDE;
            mrow[0]  = mask ? 1.0f : 0.0f;
            mrow[1]  = fb;
            mrow[2]  = (float)gic;
            mrow[3]  = (float)gjc;
            mrow[4]  = fc;
            mrow[5]  = gx - (float)gii;      // tbox.x
            mrow[6]  = gy - (float)gjj;      // tbox.y
            mrow[7]  = gw;
            mrow[8]  = gh;
            mrow[9]  = ANC_W[a];
            mrow[10] = ANC_H[a];
            mrow[11] = (float)a;
            if (mask) cnt += 1.0f;
        }
    }
    if (cnt > 0.0f) atomicAdd(nval, cnt);   // integer-valued -> exact & deterministic
}

// ---------------- WMMA K=256 microkernel ----------------------------------
// arow: per-lane A base, already adjusted by (half ? 8*KSTRIDE : 0)
// bfl : per-lane B fragment base (bfrag + lane*16), K-step stride 512 ushorts
// All in-loop addresses are compile-time immediates off these two bases.
template<int KSTRIDE>
__device__ __forceinline__ v8f wmma_k256(const float* __restrict__ arow,
                                         const unsigned short* __restrict__ bfl) {
    const int KAB[8] = {0, 2, 4, 6, 16, 18, 20, 22};
    v8f acc = {};
    #pragma unroll
    for (int ks = 0; ks < 8; ++ks) {
        union { v16bf v; unsigned short u[16]; } A;
        union { v16bf v; v4u q[2]; } B;
        const v4u* bp = (const v4u*)(bfl + ks * 512);
        B.q[0] = bp[0];
        B.q[1] = bp[1];
        #pragma unroll
        for (int vv = 0; vv < 8; ++vv) {
            const int ka = ks * 32 + KAB[vv];
            A.u[2 * vv]     = f2bf(arow[ka * KSTRIDE]);
            A.u[2 * vv + 1] = f2bf(arow[(ka + 1) * KSTRIDE]);
        }
        acc = __builtin_amdgcn_wmma_f32_16x16x32_bf16(false, A.v, false, B.v,
                                                      (short)0, acc, false, false);
    }
    return acc;
}

// ---------------- kernel 2: objectness GEMM over all positions (WMMA) ----------
// one wave per 16-row M tile; BS_*NYX is divisible by 16 -> no edge handling
template<int NYX>
__global__ void __launch_bounds__(32)
k_obj_gemm(const float* __restrict__ feat, const unsigned short* __restrict__ bfrag,
           const float* __restrict__ bias, float* __restrict__ obj) {
    const int lane = threadIdx.x;
    const int half = lane >> 4;
    const int l15  = lane & 15;
    const int mt   = blockIdx.x;

    const int p = mt * 16 + l15;
    const int b = p / NYX, rem = p % NYX;
    const float* arow = feat + (size_t)b * (INCH * NYX) + rem + (half ? 8 * NYX : 0);
    __builtin_prefetch(arow, 0, 1);                      // global_prefetch_b8
    const unsigned short* bfl = bfrag + lane * 16;

    v8f acc = wmma_k256<NYX>(arow, bfl);

    const int n = l15;
    if (n < NA_) {
        float bs = bias[n * NO_ + 4];
        #pragma unroll
        for (int r = 0; r < 8; ++r) {
            int pp = mt * 16 + half * 8 + r;
            int bb = pp / NYX, rr = pp % NYX;
            obj[((size_t)bb * NA_ + n) * NYX + rr] = acc[r] + bs;
        }
    }
}

// ---------------- kernel 3: gathered-row GEMM, all NO channels (WMMA) ----------
// grid = (Mtiles=63, Ntiles=6, anchor=3); rows grouped by anchor so B is uniform
template<int NX, int NYX>
__global__ void __launch_bounds__(32)
k_gather_gemm(const float* __restrict__ feat, const unsigned short* __restrict__ bfrag,
              const float* __restrict__ bias, const float* __restrict__ meta,
              float* __restrict__ ps) {
    const int lane = threadIdx.x;
    const int half = lane >> 4;
    const int l15  = lane & 15;
    const int mt = blockIdx.x, nt = blockIdx.y, a = blockIdx.z;

    int row = mt * 16 + l15; if (row >= ROWS_PER_A) row = ROWS_PER_A - 1;
    const float* mrow = meta + (size_t)(a * ROWS_PER_A + row) * META_STRIDE;
    const int b  = (int)mrow[1];
    const int gi = (int)mrow[2];
    const int gj = (int)mrow[3];
    const float* arow = feat + (size_t)b * (INCH * NYX) + gj * NX + gi
                             + (half ? 8 * NYX : 0);
    const unsigned short* bfl = bfrag + (size_t)((a * NTILES_O + nt) * 8) * 512
                                      + lane * 16;

    v8f acc = wmma_k256<NYX>(arow, bfl);

    const int o = nt * 16 + l15;
    if (o < NO_) {
        float bs = bias[a * NO_ + o];
        #pragma unroll
        for (int r = 0; r < 8; ++r) {
            int rowm = mt * 16 + half * 8 + r;
            if (rowm < ROWS_PER_A)
                ps[((size_t)a * ROWS_PER_A + rowm) * PS_STRIDE + o] = acc[r] + bs;
        }
    }
}

// ---------------- kernel 4: CIoU + lbox/lcls partials + tobj scatter ----------
__device__ float dev_ciou(float x1, float y1, float w1, float h1,
                          float x2, float y2, float w2, float h2) {
    const float E = 1e-9f;
    float b1x1 = x1 - w1 * 0.5f, b1x2 = x1 + w1 * 0.5f;
    float b1y1 = y1 - h1 * 0.5f, b1y2 = y1 + h1 * 0.5f;
    float b2x1 = x2 - w2 * 0.5f, b2x2 = x2 + w2 * 0.5f;
    float b2y1 = y2 - h2 * 0.5f, b2y2 = y2 + h2 * 0.5f;
    float iw = fmaxf(fminf(b1x2, b2x2) - fmaxf(b1x1, b2x1), 0.0f);
    float ih = fmaxf(fminf(b1y2, b2y2) - fmaxf(b1y1, b2y1), 0.0f);
    float inter = iw * ih;
    float uni = w1 * h1 + w2 * h2 - inter + E;
    float iou = inter / uni;
    float cw = fmaxf(b1x2, b2x2) - fminf(b1x1, b2x1);
    float ch = fmaxf(b1y2, b2y2) - fminf(b1y1, b2y1);
    float c2 = cw * cw + ch * ch + E;
    float dx = b2x1 + b2x2 - b1x1 - b1x2;
    float dy = b2y1 + b2y2 - b1y1 - b1y2;
    float rho2 = (dx * dx + dy * dy) * 0.25f;
    float dv = atanf(w2 / (h2 + E)) - atanf(w1 / (h1 + E));
    float v = (4.0f / 9.8696044f) * dv * dv;
    float alpha = v / (v - iou + 1.0f + E);
    return iou - (rho2 / c2 + v * alpha);
}

__global__ void k_scatter_loss(const float* __restrict__ meta, const float* __restrict__ ps,
                               unsigned* __restrict__ tobj, const float* __restrict__ nvalp,
                               float* __restrict__ lboxp, float* __restrict__ lclsp,
                               int nx, int nyx) {
    __shared__ float sb[256], sc[256];
    int tid = blockIdx.x * blockDim.x + threadIdx.x;
    int stride = gridDim.x * blockDim.x;
    float nval = *nvalp;
    float pw_c = nval * (float)(NC_ - 1) * 0.5f / fmaxf(nval, 2.0f);
    float lb = 0.0f, lc = 0.0f;
    for (int row = tid; row < ROWS_LVL; row += stride) {
        const float* m = meta + (size_t)row * META_STRIDE;
        float mask = m[0];
        int b = (int)m[1], gi = (int)m[2], gj = (int)m[3], cls = (int)m[4];
        int a = row / ROWS_PER_A;
        const float* p = ps + (size_t)row * PS_STRIDE;
        float px = dsigmoid(p[0]) * 2.0f - 0.5f;
        float py = dsigmoid(p[1]) * 2.0f - 0.5f;
        float sw = dsigmoid(p[2]) * 2.0f;
        float sh = dsigmoid(p[3]) * 2.0f;
        float pw = sw * sw * m[9];
        float ph = sh * sh * m[10];
        float iou = dev_ciou(px, py, pw, ph, m[5], m[6], m[7], m[8]);
        lb += mask * (1.0f - iou);
        if (mask > 0.5f) {
            unsigned bits = __float_as_uint(fmaxf(iou, 0.0f));
            atomicMax(tobj + ((size_t)b * NA_ + a) * nyx + (size_t)gj * nx + gi, bits);
        }
        float cl = 0.0f;
        for (int c = 0; c < NC_; ++c) {
            float l = p[5 + c];
            float t = (c == cls) ? 1.0f : 0.0f;
            cl += pw_c * t * dsoftplus(-l) + (1.0f - t) * dsoftplus(l);
        }
        lc += mask * cl;
    }
    sb[threadIdx.x] = lb; sc[threadIdx.x] = lc;
    __syncthreads();
    for (int s = 128; s > 0; s >>= 1) {
        if ((int)threadIdx.x < s) {
            sb[threadIdx.x] += sb[threadIdx.x + s];
            sc[threadIdx.x] += sc[threadIdx.x + s];
        }
        __syncthreads();
    }
    if (threadIdx.x == 0) { lboxp[blockIdx.x] = sb[0]; lclsp[blockIdx.x] = sc[0]; }
}

// ---------------- kernel 5: fg/bg counts over tobj ----------------
__global__ void k_counts(const unsigned* __restrict__ tobj, float* __restrict__ fgp,
                         float* __restrict__ bgp, int E) {
    __shared__ float sf[256], sg[256];
    int tid = blockIdx.x * blockDim.x + threadIdx.x;
    int stride = gridDim.x * blockDim.x;
    float fg = 0.0f, bg = 0.0f;
    for (int i = tid; i < E; i += stride) {
        unsigned v = tobj[i];
        fg += (v != 0u) ? 1.0f : 0.0f;
        bg += (v == 0u) ? 1.0f : 0.0f;
    }
    sf[threadIdx.x] = fg; sg[threadIdx.x] = bg;
    __syncthreads();
    for (int s = 128; s > 0; s >>= 1) {
        if ((int)threadIdx.x < s) {
            sf[threadIdx.x] += sf[threadIdx.x + s];
            sg[threadIdx.x] += sg[threadIdx.x + s];
        }
        __syncthreads();
    }
    if (threadIdx.x == 0) { fgp[blockIdx.x] = sf[0]; bgp[blockIdx.x] = sg[0]; }
}

// ---------------- kernel 6: objectness BCE partials ----------------
__global__ void k_obj_bce(const float* __restrict__ obj, const unsigned* __restrict__ tobj,
                          const float* __restrict__ fgp, const float* __restrict__ bgp,
                          float* __restrict__ bcep, int E) {
    __shared__ float sr[256];
    __shared__ float s_pwo;
    if (threadIdx.x == 0) {
        float fg = 0.0f, bg = 0.0f;
        for (int i = 0; i < 128; ++i) { fg += fgp[i]; bg += bgp[i]; }
        s_pwo = bg * 0.5f / fmaxf(fg, 2.0f);
    }
    __syncthreads();
    float pw = s_pwo;
    int tid = blockIdx.x * blockDim.x + threadIdx.x;
    int stride = gridDim.x * blockDim.x;
    float acc = 0.0f;
    for (int i = tid; i < E; i += stride) {
        float l = obj[i];
        float t = __uint_as_float(tobj[i]);
        acc += pw * t * dsoftplus(-l) + (1.0f - t) * dsoftplus(l);
    }
    sr[threadIdx.x] = acc;
    __syncthreads();
    for (int s = 128; s > 0; s >>= 1) {
        if ((int)threadIdx.x < s) sr[threadIdx.x] += sr[threadIdx.x + s];
        __syncthreads();
    }
    if (threadIdx.x == 0) bcep[blockIdx.x] = sr[0];
}

// ---------------- kernel 7: finalize (single thread, fixed order) ----------------
__global__ void k_finalize(const float* __restrict__ ws, float* __restrict__ out) {
    if (threadIdx.x != 0 || blockIdx.x != 0) return;
    const float bal[3] = {4.0f, 1.0f, 0.25f};
    const int   Ev[3]  = {153600, 38400, 9600};
    float lbox = 0.0f, lobj = 0.0f, lcls = 0.0f;
    for (int lvl = 0; lvl < 3; ++lvl) {
        float nv = ws[WS_NVAL + lvl];
        float sb = 0.0f, sc = 0.0f, so = 0.0f;
        for (int i = 0; i < 16; ++i)  sb += ws[WS_LBOXP + lvl * 16 + i];
        for (int i = 0; i < 16; ++i)  sc += ws[WS_LCLSP + lvl * 16 + i];
        for (int i = 0; i < 128; ++i) so += ws[WS_BCEP + lvl * 128 + i];
        lbox += sb / fmaxf(nv, 1.0f);
        lcls += sc / fmaxf(nv * (float)NC_, 1.0f);
        lobj += (so / (float)Ev[lvl]) * bal[lvl];
    }
    out[0] = lbox * 0.05f;
    out[1] = lobj * 1.0f;
    out[2] = lcls * 0.5f;
}

// ---------------- host launcher ----------------
extern "C" void kernel_launch(void* const* d_in, const int* in_sizes, int n_in,
                              void* d_out, int out_size, void* d_ws, size_t ws_size,
                              hipStream_t stream) {
    (void)in_sizes; (void)n_in; (void)out_size;
    if (ws_size < (size_t)WS_TOTAL * sizeof(float)) return;

    const float* feat[3] = {(const float*)d_in[0], (const float*)d_in[1], (const float*)d_in[2]};
    const float* wm[3]   = {(const float*)d_in[3], (const float*)d_in[4], (const float*)d_in[5]};
    const float* bi[3]   = {(const float*)d_in[6], (const float*)d_in[7], (const float*)d_in[8]};
    const float* targets = (const float*)d_in[9];
    float* ws = (float*)d_ws;

    const int NX[3]  = {80, 40, 20};
    const int NYX[3] = {6400, 1600, 400};
    const int OBJ_OFS[3] = {0, 153600, 192000};
    const int Ev[3] = {153600, 38400, 9600};

    k_zero<<<512, 256, 0, stream>>>(ws);

    for (int lvl = 0; lvl < 3; ++lvl) {
        k_pack_obj_b<<<1, 256, 0, stream>>>(
            wm[lvl], (unsigned short*)(ws + WS_BF_OBJ) + (size_t)lvl * 4096);
        k_pack_gat_b<<<18, 256, 0, stream>>>(
            wm[lvl], (unsigned short*)(ws + WS_BF_GAT) + (size_t)lvl * 73728);
        k_build_targets<<<3, 256, 0, stream>>>(
            targets, ws + WS_META + lvl * (ROWS_LVL * META_STRIDE),
            ws + WS_NVAL + lvl, NX[lvl]);
    }

    // objectness GEMMs (compile-time geometry per level)
    {
        const unsigned short* bo0 = (const unsigned short*)(ws + WS_BF_OBJ);
        k_obj_gemm<6400><<<BS_ * 6400 / 16, 32, 0, stream>>>(
            feat[0], bo0 + 0 * 4096, bi[0], ws + WS_OBJ + OBJ_OFS[0]);
        k_obj_gemm<1600><<<BS_ * 1600 / 16, 32, 0, stream>>>(
            feat[1], bo0 + 1 * 4096, bi[1], ws + WS_OBJ + OBJ_OFS[1]);
        k_obj_gemm< 400><<<BS_ *  400 / 16, 32, 0, stream>>>(
            feat[2], bo0 + 2 * 4096, bi[2], ws + WS_OBJ + OBJ_OFS[2]);
    }
    // gathered-row GEMMs
    {
        const unsigned short* bg0 = (const unsigned short*)(ws + WS_BF_GAT);
        dim3 grid((ROWS_PER_A + 15) / 16, NTILES_O, NA_);
        k_gather_gemm<80, 6400><<<grid, 32, 0, stream>>>(
            feat[0], bg0 + 0 * 73728, bi[0],
            ws + WS_META + 0 * (ROWS_LVL * META_STRIDE),
            ws + WS_PS + 0 * (ROWS_LVL * PS_STRIDE));
        k_gather_gemm<40, 1600><<<grid, 32, 0, stream>>>(
            feat[1], bg0 + 1 * 73728, bi[1],
            ws + WS_META + 1 * (ROWS_LVL * META_STRIDE),
            ws + WS_PS + 1 * (ROWS_LVL * PS_STRIDE));
        k_gather_gemm<20, 400><<<grid, 32, 0, stream>>>(
            feat[2], bg0 + 2 * 73728, bi[2],
            ws + WS_META + 2 * (ROWS_LVL * META_STRIDE),
            ws + WS_PS + 2 * (ROWS_LVL * PS_STRIDE));
    }

    for (int lvl = 0; lvl < 3; ++lvl) {
        k_scatter_loss<<<16, 256, 0, stream>>>(
            ws + WS_META + lvl * (ROWS_LVL * META_STRIDE),
            ws + WS_PS + lvl * (ROWS_LVL * PS_STRIDE),
            (unsigned*)(ws + WS_TOBJ) + OBJ_OFS[lvl],
            ws + WS_NVAL + lvl,
            ws + WS_LBOXP + lvl * 16, ws + WS_LCLSP + lvl * 16,
            NX[lvl], NYX[lvl]);
    }
    for (int lvl = 0; lvl < 3; ++lvl) {
        k_counts<<<128, 256, 0, stream>>>(
            (const unsigned*)(ws + WS_TOBJ) + OBJ_OFS[lvl],
            ws + WS_FGP + lvl * 128, ws + WS_BGP + lvl * 128, Ev[lvl]);
    }
    for (int lvl = 0; lvl < 3; ++lvl) {
        k_obj_bce<<<128, 256, 0, stream>>>(
            ws + WS_OBJ + OBJ_OFS[lvl],
            (const unsigned*)(ws + WS_TOBJ) + OBJ_OFS[lvl],
            ws + WS_FGP + lvl * 128, ws + WS_BGP + lvl * 128,
            ws + WS_BCEP + lvl * 128, Ev[lvl]);
    }
    k_finalize<<<1, 32, 0, stream>>>(ws, (float*)d_out);
}
